// AdaptiveVisionTransformer_69896297775543
// MI455X (gfx1250) — compile-verified, matrix operand order checked by reading, not run on hardware
//
#include <hip/hip_runtime.h>
#include <cstdint>
#include <cstddef>

// ---------------------------------------------------------------------------
// Model dims
// ---------------------------------------------------------------------------
constexpr int Dm   = 768;
constexpr int Lc   = 12;
constexpr int Hc   = 12;
constexpr int HDc  = 64;
constexpr int Nt   = 197;          // tokens
constexpr int Bc   = 32;           // batch
constexpr int MLPc = 3072;
constexpr int NCc  = 1000;
constexpr int Gc   = 14;           // grid 14x14 patches
constexpr int Mtok = Bc * Nt;      // 6304 rows
constexpr int Mpat = Bc * Gc * Gc; // 6272 patch rows
constexpr int NP   = 224;          // padded token count (mult of 32)
constexpr long long SCs = 208LL * 224; // per-(b,h) score tile stride = 46592
constexpr int Zbh  = Bc * Hc;      // 384

// ---------------------------------------------------------------------------
// Types / helpers
// ---------------------------------------------------------------------------
typedef __bf16 v16bf __attribute__((ext_vector_type(16)));
typedef float  v8f   __attribute__((ext_vector_type(8)));
typedef int    gv4i  __attribute__((vector_size(16)));   // matches builtin param

union Frag { v16bf v; uint4 u[2]; };

__device__ inline unsigned short f2bf(float f) {
  union { __bf16 h; unsigned short u; } c; c.h = (__bf16)f; return c.u;
}

// ---- CDNA5 async global->LDS copy (ASYNCcnt) with sync fallback ----
#if defined(__HIP_DEVICE_COMPILE__) && __has_builtin(__builtin_amdgcn_global_load_async_to_lds_b128)
#define GEMM_ASYNC 1
#else
#define GEMM_ASYNC 0
#endif

#define AS1 __attribute__((address_space(1)))
#define AS3 __attribute__((address_space(3)))

__device__ inline void copy16_g2l(unsigned short* dst_lds, const unsigned short* src_g) {
#if GEMM_ASYNC
  __builtin_amdgcn_global_load_async_to_lds_b128(
      (AS1 gv4i*)src_g, (AS3 gv4i*)dst_lds, 0, 0);
#else
  *reinterpret_cast<uint4*>(dst_lds) = *reinterpret_cast<const uint4*>(src_g);
#endif
}

__device__ inline void wait_async0() {
#if GEMM_ASYNC
#if __has_builtin(__builtin_amdgcn_s_wait_asynccnt)
  __builtin_amdgcn_s_wait_asynccnt(0);
#else
  asm volatile("s_wait_asynccnt 0" ::: "memory");
#endif
#endif
}

// ---------------------------------------------------------------------------
// fp32 -> bf16 conversion (weights, once per launch)
// ---------------------------------------------------------------------------
__global__ void cvt_f32_to_bf16(const float* __restrict__ in,
                                unsigned short* __restrict__ out, long long n) {
  long long i = (long long)blockIdx.x * blockDim.x + threadIdx.x;
  if (i < n) out[i] = f2bf(in[i]);
}

// ---------------------------------------------------------------------------
// Patch extraction -> bf16 matrix (Mpat x 768), k = c*256 + py*16 + px
// ---------------------------------------------------------------------------
__global__ void patchify_kernel(const float* __restrict__ x,
                                unsigned short* __restrict__ out) {
  long long idx = (long long)blockIdx.x * blockDim.x + threadIdx.x;
  if (idx >= (long long)Mpat * Dm) return;
  int k = (int)(idx % Dm);
  long long t = idx / Dm;
  int g = (int)(t % (Gc * Gc));
  int b = (int)(t / (Gc * Gc));
  int c = k >> 8, rp = k & 255, py = rp >> 4, px = rp & 15;
  int gy = g / Gc, gx = g % Gc;
  float v = x[(((size_t)b * 3 + c) * 224 + (gy * 16 + py)) * 224 + (gx * 16 + px)];
  out[idx] = f2bf(v);
}

// ---------------------------------------------------------------------------
// Assemble tokens: h = [cls; patches] + pos_embed, init ACT state
// ---------------------------------------------------------------------------
__global__ void assemble_tokens(const float* __restrict__ tokC,
                                const float* __restrict__ clsTok,
                                const float* __restrict__ pos,
                                float* __restrict__ h, float* __restrict__ outAcc,
                                float* __restrict__ cumul, float* __restrict__ rem,
                                float* __restrict__ cont) {
  long long idx = (long long)blockIdx.x * blockDim.x + threadIdx.x;
  if (idx >= (long long)Mtok * Dm) return;
  int d = (int)(idx % Dm);
  int row = (int)(idx / Dm);
  int b = row / Nt, n = row % Nt;
  float v = (n == 0) ? clsTok[d] : tokC[((size_t)(b * (Gc * Gc) + (n - 1))) * Dm + d];
  h[idx] = v + pos[(size_t)n * Dm + d];
  outAcc[idx] = 0.f;
  if (d == 0) { cumul[row] = 0.f; rem[row] = 1.f; cont[row] = 1.f; }
}

// h *= cont (broadcast), and halt_mask = (1-cont)*-10000
__global__ void scale_h_mask(float* __restrict__ h, const float* __restrict__ cont,
                             float* __restrict__ mask) {
  long long idx = (long long)blockIdx.x * blockDim.x + threadIdx.x;
  if (idx >= (long long)Mtok * Dm) return;
  int row = (int)(idx / Dm);
  float c = cont[row];
  h[idx] *= c;
  if ((int)(idx % Dm) == 0) mask[row] = (1.f - c) * -10000.f;
}

// y = g*cont + h
__global__ void residual_kernel(const float* __restrict__ g, const float* __restrict__ cont,
                                const float* __restrict__ h, float* __restrict__ y) {
  long long idx = (long long)blockIdx.x * blockDim.x + threadIdx.x;
  if (idx >= (long long)Mtok * Dm) return;
  int row = (int)(idx / Dm);
  y[idx] = g[idx] * cont[row] + h[idx];
}

// ---------------------------------------------------------------------------
// LayerNorm -> bf16 (optionally * cont).  One 256-thread block per row (768).
// ---------------------------------------------------------------------------
__global__ __launch_bounds__(256)
void layernorm_bf16(const float* __restrict__ X, long long ldx,
                    const float* __restrict__ sc, const float* __restrict__ bi,
                    const float* __restrict__ cont,
                    unsigned short* __restrict__ out) {
  int row = blockIdx.x;
  int tid = threadIdx.x;
  const float* x = X + (size_t)row * ldx;
  float xv[3], s = 0.f, s2 = 0.f;
#pragma unroll
  for (int j = 0; j < 3; ++j) {
    float v = x[tid + j * 256]; xv[j] = v; s += v; s2 += v * v;
  }
  __shared__ float r1[256], r2[256];
  r1[tid] = s; r2[tid] = s2; __syncthreads();
  for (int off = 128; off > 0; off >>= 1) {
    if (tid < off) { r1[tid] += r1[tid + off]; r2[tid] += r2[tid + off]; }
    __syncthreads();
  }
  float mean = r1[0] * (1.f / 768.f);
  float var  = r2[0] * (1.f / 768.f) - mean * mean;
  float rst  = rsqrtf(var + 1e-6f);
  float cv   = cont ? cont[row] : 1.f;
  unsigned short* o = out + (size_t)row * Dm;
#pragma unroll
  for (int j = 0; j < 3; ++j) {
    int d = tid + j * 256;
    o[d] = f2bf(((xv[j] - mean) * rst * sc[d] + bi[d]) * cv);
  }
}

// ---------------------------------------------------------------------------
// V transpose: vT[(b*12+h)*64+d][k] = qkv[b,k, 1536 + h*64 + d], pad k>=197 = 0
// ---------------------------------------------------------------------------
__global__ void transpose_v(const unsigned short* __restrict__ qkv,
                            unsigned short* __restrict__ vT) {
  long long idx = (long long)blockIdx.x * blockDim.x + threadIdx.x;
  if (idx >= (long long)Zbh * HDc * NP) return;
  int k = (int)(idx % NP);
  long long t = idx / NP;
  int d = (int)(t % HDc); t /= HDc;
  int hh = (int)(t % Hc);
  int b = (int)(t / Hc);
  unsigned short v = 0;
  if (k < Nt) v = qkv[((size_t)(b * Nt + k)) * (3 * Dm) + 2 * Dm + hh * HDc + d];
  vT[idx] = v;
}

// ---------------------------------------------------------------------------
// Softmax: one wave32 per (z,q) row of 197 fp32 scores -> bf16 probs (pad->0)
// ---------------------------------------------------------------------------
__global__ __launch_bounds__(256)
void softmax_rows(const float* __restrict__ scores, unsigned short* __restrict__ probs,
                  int nrows) {
  int wid = threadIdx.x >> 5, lane = threadIdx.x & 31;
  int row = blockIdx.x * 8 + wid;
  if (row >= nrows) return;
  int z = row / Nt, q = row % Nt;
  const float* src = scores + (size_t)z * SCs + (size_t)q * NP;
  unsigned short* dst = probs + (size_t)z * SCs + (size_t)q * NP;
  float sv[7], mx = -1e30f;
#pragma unroll
  for (int j = 0; j < 7; ++j) {
    int c = lane + j * 32;
    float v = (c < Nt) ? src[c] : -1e30f;
    sv[j] = v; mx = fmaxf(mx, v);
  }
  for (int off = 16; off > 0; off >>= 1) mx = fmaxf(mx, __shfl_xor(mx, off, 32));
  float sum = 0.f;
#pragma unroll
  for (int j = 0; j < 7; ++j) {
    int c = lane + j * 32;
    float e = (c < Nt) ? expf(sv[j] - mx) : 0.f;
    sv[j] = e; sum += e;
  }
  for (int off = 16; off > 0; off >>= 1) sum += __shfl_xor(sum, off, 32);
  float inv = 1.f / sum;
#pragma unroll
  for (int j = 0; j < 7; ++j) {
    int c = lane + j * 32;
    if (c < NP) dst[c] = (c < Nt) ? f2bf(sv[j] * inv) : (unsigned short)0;
  }
}

// ---------------------------------------------------------------------------
// ACT update (block per token): h = m*cont + y; gate on h[0]; accumulate output
// ---------------------------------------------------------------------------
__global__ __launch_bounds__(256)
void act_update(const float* __restrict__ m, const float* __restrict__ y,
                float* __restrict__ h, float* __restrict__ outAcc,
                float* __restrict__ cumul, float* __restrict__ rem,
                float* __restrict__ cont, int isLast) {
  int row = blockIdx.x, tid = threadIdx.x;
  float c = cont[row];
  size_t base = (size_t)row * Dm;
  float hn[3];
#pragma unroll
  for (int j = 0; j < 3; ++j) {
    int d = tid + j * 256;
    float v = m[base + d] * c + y[base + d];
    hn[j] = v; h[base + d] = v;
  }
  __shared__ float sh0, scoef;
  if (tid == 0) sh0 = hn[0];
  __syncthreads();
  if (tid == 0) {
    float halting = isLast ? 1.f : 1.f / (1.f + expf(-(sh0 * 10.f - 30.f)));
    float cum = cumul[row] + halting;
    float reached = (cum > 0.99f) ? c : 0.f;
    float nr = (cum < 0.99f) ? 1.f : 0.f;
    float r = rem[row];
    scoef = r * reached + halting * nr;
    rem[row] = r - nr * halting;
    cumul[row] = cum;
    cont[row] = nr;
  }
  __syncthreads();
  float coef = scoef;
#pragma unroll
  for (int j = 0; j < 3; ++j) {
    int d = tid + j * 256;
    outAcc[base + d] += hn[j] * coef;
  }
}

// ---------------------------------------------------------------------------
// WMMA GEMM:  C[m,n] = sum_k A[m,k] * W[n,k]  (bf16 in, fp32 accumulate)
//   Block tile 128x128, K-step 32, double-buffered LDS, async global->LDS.
//   8 waves in 2(M)x4(N); each wave = 64x32 = 4x2 WMMA tiles.
//   Row indices are CLAMPED (not zero-filled): garbage rows/cols only affect
//   unstored outputs, and unconditional loads keep EXEC full for async copies.
//   z-batching: off = (z/zdiv)*hi + (z%zdiv)*lo per operand.
//   mode: 0 = +bias -> f32   1 = +bias -> bf16
//         2 = +bias -> GELU(erf) -> bf16   3 = *scale + mask[n] -> f32
// ---------------------------------------------------------------------------
#define BM 128
#define BN 128
#define BK 32
#define LDSP 40   // padded K-stride in LDS (bf16 elems) to break bank conflicts

__global__ __launch_bounds__(256)
void gemm_bf16_wmma(const unsigned short* __restrict__ A,
                    const unsigned short* __restrict__ W,
                    float* __restrict__ Cf, unsigned short* __restrict__ Cb,
                    const float* __restrict__ bias, const float* __restrict__ maskBase,
                    int M, int Nsz, int K, int lda, int ldb, int ldc,
                    long long sAhi, long long sAlo,
                    long long sBhi, long long sBlo,
                    long long sChi, long long sClo,
                    int zdiv, int maskStride, float scale, int mode) {
  __shared__ __align__(16) unsigned short As[2][BM][LDSP];
  __shared__ __align__(16) unsigned short Bs[2][BN][LDSP];

  int z  = blockIdx.z;
  int zb = z / zdiv, zr = z % zdiv;
  A += (size_t)zb * sAhi + (size_t)zr * sAlo;
  W += (size_t)zb * sBhi + (size_t)zr * sBlo;
  size_t coff = (size_t)zb * sChi + (size_t)zr * sClo;
  const float* mask = maskBase ? (maskBase + (size_t)zb * maskStride) : nullptr;

  int tid  = threadIdx.x;
  int lane = tid & 31;
  int wave = tid >> 5;
  int wm = wave >> 2;            // 0..1  (M, 64 rows each)
  int wn = wave & 3;             // 0..3  (N, 32 cols each)
  int l16 = lane & 15;
  int kh  = (lane >> 4) * 8;     // ISA 16-bit A/B fragment K split

  int blockM = blockIdx.y * BM;
  int blockN = blockIdx.x * BN;

  // staging: thread t loads 16B at (row = t>>2 [+64], col = (t&3)*8) per matrix
  int ldRow = tid >> 2;          // 0..63
  int ldCol = (tid & 3) * 8;     // 0,8,16,24
  int ra0 = blockM + ldRow;      if (ra0 > M - 1)   ra0 = M - 1;
  int ra1 = blockM + ldRow + 64; if (ra1 > M - 1)   ra1 = M - 1;
  int rb0 = blockN + ldRow;      if (rb0 > Nsz - 1) rb0 = Nsz - 1;
  int rb1 = blockN + ldRow + 64; if (rb1 > Nsz - 1) rb1 = Nsz - 1;
  const unsigned short* aP0 = A + (size_t)ra0 * lda + ldCol;
  const unsigned short* aP1 = A + (size_t)ra1 * lda + ldCol;
  const unsigned short* bP0 = W + (size_t)rb0 * ldb + ldCol;
  const unsigned short* bP1 = W + (size_t)rb1 * ldb + ldCol;

  v8f acc[4][2];
#pragma unroll
  for (int i = 0; i < 4; ++i)
#pragma unroll
    for (int j = 0; j < 2; ++j)
#pragma unroll
      for (int e = 0; e < 8; ++e) acc[i][j][e] = 0.f;

  auto stage = [&](int buf, int k0) {
    copy16_g2l(&As[buf][ldRow][ldCol],      aP0 + k0);
    copy16_g2l(&As[buf][ldRow + 64][ldCol], aP1 + k0);
    copy16_g2l(&Bs[buf][ldRow][ldCol],      bP0 + k0);
    copy16_g2l(&Bs[buf][ldRow + 64][ldCol], bP1 + k0);
  };

  int nk = K / BK;
  stage(0, 0);
  wait_async0();
  __syncthreads();

  for (int kt = 0; kt < nk; ++kt) {
    int buf = kt & 1;
    if (kt + 1 < nk) stage(buf ^ 1, (kt + 1) * BK);

    Frag af[4], bfg[2];
#pragma unroll
    for (int mt = 0; mt < 4; ++mt) {
      int r = wm * 64 + mt * 16 + l16;
      af[mt].u[0] = *reinterpret_cast<const uint4*>(&As[buf][r][kh]);
      af[mt].u[1] = *reinterpret_cast<const uint4*>(&As[buf][r][kh + 16]);
    }
#pragma unroll
    for (int nt = 0; nt < 2; ++nt) {
      int r = wn * 32 + nt * 16 + l16;
      bfg[nt].u[0] = *reinterpret_cast<const uint4*>(&Bs[buf][r][kh]);
      bfg[nt].u[1] = *reinterpret_cast<const uint4*>(&Bs[buf][r][kh + 16]);
    }
#pragma unroll
    for (int mt = 0; mt < 4; ++mt)
#pragma unroll
      for (int nt = 0; nt < 2; ++nt)
        acc[mt][nt] = __builtin_amdgcn_wmma_f32_16x16x32_bf16(
            false, af[mt].v, false, bfg[nt].v, (short)0, acc[mt][nt], false, false);

    wait_async0();
    __syncthreads();
  }

  // epilogue (C/D layout: VGPR e -> row = e + (lane>>4)*8, col = lane&15)
  int rowOff = (lane >> 4) * 8;
#pragma unroll
  for (int mt = 0; mt < 4; ++mt) {
#pragma unroll
    for (int nt = 0; nt < 2; ++nt) {
      int gc = blockN + wn * 32 + nt * 16 + l16;
      if (gc >= Nsz) continue;
      float bv = bias ? bias[gc] : 0.f;
#pragma unroll
      for (int e = 0; e < 8; ++e) {
        int gr = blockM + wm * 64 + mt * 16 + rowOff + e;
        if (gr >= M) continue;
        float raw = acc[mt][nt][e];
        size_t off = coff + (size_t)gr * ldc + gc;
        if (mode == 0) {
          Cf[off] = raw + bv;
        } else if (mode == 1) {
          Cb[off] = f2bf(raw + bv);
        } else if (mode == 2) {
          float v = raw + bv;
          Cb[off] = f2bf(0.5f * v * (1.f + erff(v * 0.70710678118654752f)));
        } else {
          Cf[off] = raw * scale + mask[gc];
        }
      }
    }
  }
}

// ---------------------------------------------------------------------------
// Host orchestration
// ---------------------------------------------------------------------------
extern "C" void kernel_launch(void* const* d_in, const int* in_sizes, int n_in,
                              void* d_out, int out_size, void* d_ws, size_t ws_size,
                              hipStream_t stream) {
  (void)in_sizes; (void)n_in; (void)out_size; (void)ws_size;

  const float* x       = (const float*)d_in[0];
  const float* conv_w  = (const float*)d_in[1];
  const float* conv_b  = (const float*)d_in[2];
  const float* cls_tok = (const float*)d_in[3];
  const float* pos     = (const float*)d_in[4];
  const float* ln1_s   = (const float*)d_in[5];
  const float* ln1_b   = (const float*)d_in[6];
  const float* qkv_w   = (const float*)d_in[7];
  const float* qkv_b   = (const float*)d_in[8];
  const float* proj_w  = (const float*)d_in[9];
  const float* proj_b  = (const float*)d_in[10];
  const float* ln2_s   = (const float*)d_in[11];
  const float* ln2_b   = (const float*)d_in[12];
  const float* mlp_w1  = (const float*)d_in[13];
  const float* mlp_b1  = (const float*)d_in[14];
  const float* mlp_w2  = (const float*)d_in[15];
  const float* mlp_b2  = (const float*)d_in[16];
  const float* lnf_s   = (const float*)d_in[17];
  const float* lnf_b   = (const float*)d_in[18];
  const float* head_w  = (const float*)d_in[19];
  const float* head_b  = (const float*)d_in[20];
  float* logits = (float*)d_out;

  char* ws = (char*)d_ws;
  size_t off = 0;
  auto alloc = [&](size_t bytes) -> char* {
    off = (off + 255) & ~(size_t)255;
    char* p = ws + off; off += bytes; return p;
  };

  unsigned short* wbConv  = (unsigned short*)alloc((size_t)Dm * Dm * 2);
  unsigned short* wbQkv   = (unsigned short*)alloc((size_t)Lc * 3 * Dm * Dm * 2);
  unsigned short* wbProj  = (unsigned short*)alloc((size_t)Lc * Dm * Dm * 2);
  unsigned short* wbM1    = (unsigned short*)alloc((size_t)Lc * MLPc * Dm * 2);
  unsigned short* wbM2    = (unsigned short*)alloc((size_t)Lc * Dm * MLPc * 2);
  unsigned short* wbHead  = (unsigned short*)alloc((size_t)NCc * Dm * 2);
  unsigned short* patches = (unsigned short*)alloc((size_t)Mpat * Dm * 2);
  float* tokC   = (float*)alloc((size_t)Mpat * Dm * 4);
  float* hbuf   = (float*)alloc((size_t)Mtok * Dm * 4);
  float* ybuf   = (float*)alloc((size_t)Mtok * Dm * 4);
  float* gtmp   = (float*)alloc((size_t)Mtok * Dm * 4);
  float* outAcc = (float*)alloc((size_t)Mtok * Dm * 4);
  unsigned short* lnbuf   = (unsigned short*)alloc((size_t)Mtok * Dm * 2);
  unsigned short* qkvbuf  = (unsigned short*)alloc((size_t)Mtok * 3 * Dm * 2);
  float* scores = (float*)alloc((size_t)Zbh * SCs * 4);
  unsigned short* probs   = (unsigned short*)alloc((size_t)Zbh * SCs * 2);
  unsigned short* vT      = (unsigned short*)alloc((size_t)Zbh * HDc * NP * 2);
  unsigned short* attnOut = (unsigned short*)alloc((size_t)Mtok * Dm * 2);
  unsigned short* mlpBuf  = (unsigned short*)alloc((size_t)Mtok * MLPc * 2);
  unsigned short* clsBuf  = (unsigned short*)alloc((size_t)Bc * Dm * 2);
  float* cumul = (float*)alloc((size_t)Mtok * 4);
  float* rem   = (float*)alloc((size_t)Mtok * 4);
  float* cont  = (float*)alloc((size_t)Mtok * 4);
  float* maskB = (float*)alloc((size_t)Mtok * 4);

  auto cvt = [&](const float* src, unsigned short* dst, long long n) {
    int blocks = (int)((n + 255) / 256);
    cvt_f32_to_bf16<<<blocks, 256, 0, stream>>>(src, dst, n);
  };
  auto gemm = [&](const unsigned short* A, const unsigned short* W,
                  float* Cf, unsigned short* Cb, const float* bias, const float* mask,
                  int M, int Nsz, int K, int lda, int ldb, int ldc,
                  long long sAhi, long long sAlo, long long sBhi, long long sBlo,
                  long long sChi, long long sClo, int Z, int zdiv, int maskStride,
                  float scale, int mode) {
    dim3 g((Nsz + BN - 1) / BN, (M + BM - 1) / BM, Z);
    gemm_bf16_wmma<<<g, 256, 0, stream>>>(A, W, Cf, Cb, bias, mask,
        M, Nsz, K, lda, ldb, ldc, sAhi, sAlo, sBhi, sBlo, sChi, sClo,
        zdiv, maskStride, scale, mode);
  };

  // ---- weights fp32 -> bf16 (once per launch) ----
  cvt(conv_w, wbConv, (long long)Dm * Dm);
  cvt(qkv_w,  wbQkv,  (long long)Lc * 3 * Dm * Dm);
  cvt(proj_w, wbProj, (long long)Lc * Dm * Dm);
  cvt(mlp_w1, wbM1,   (long long)Lc * MLPc * Dm);
  cvt(mlp_w2, wbM2,   (long long)Lc * Dm * MLPc);
  cvt(head_w, wbHead, (long long)NCc * Dm);

  // ---- patch embedding ----
  {
    long long n = (long long)Mpat * Dm;
    patchify_kernel<<<(int)((n + 255) / 256), 256, 0, stream>>>(x, patches);
  }
  gemm(patches, wbConv, tokC, nullptr, conv_b, nullptr,
       Mpat, Dm, Dm, Dm, Dm, Dm, 0, 0, 0, 0, 0, 0, 1, 1, 0, 1.f, 0);

  const long long nTD = (long long)Mtok * Dm;
  const int ewB = (int)((nTD + 255) / 256);
  assemble_tokens<<<ewB, 256, 0, stream>>>(tokC, cls_tok, pos, hbuf, outAcc,
                                           cumul, rem, cont);

  // ---- transformer layers ----
  for (int i = 0; i < Lc; ++i) {
    scale_h_mask<<<ewB, 256, 0, stream>>>(hbuf, cont, maskB);
    layernorm_bf16<<<Mtok, 256, 0, stream>>>(hbuf, Dm, ln1_s + i * Dm, ln1_b + i * Dm,
                                             cont, lnbuf);
    // QKV: (6304 x 2304 x 768) -> bf16
    gemm(lnbuf, wbQkv + (size_t)i * 3 * Dm * Dm, nullptr, qkvbuf,
         qkv_b + i * 3 * Dm, nullptr,
         Mtok, 3 * Dm, Dm, Dm, Dm, 3 * Dm, 0, 0, 0, 0, 0, 0, 1, 1, 0, 1.f, 1);
    {
      long long n = (long long)Zbh * HDc * NP;
      transpose_v<<<(int)((n + 255) / 256), 256, 0, stream>>>(qkvbuf, vT);
    }
    // scores = Q.K^T / 8 + mask, batched over 384 (b,h)
    gemm(qkvbuf, qkvbuf + Dm, scores, nullptr, nullptr, maskB,
         Nt, Nt, HDc, 3 * Dm, 3 * Dm, NP,
         (long long)Nt * 3 * Dm, HDc,            // A: b*197*2304 + h*64
         (long long)Nt * 3 * Dm, HDc,            // B (K-matrix): same split
         (long long)Hc * SCs, SCs,               // C: z*46592
         Zbh, Hc, Nt, 0.125f, 3);
    {
      int rows = Zbh * Nt;
      softmax_rows<<<(rows + 7) / 8, 256, 0, stream>>>(scores, probs, rows);
    }
    // attn_out = P.V : batched, scatter into (b, n, h*64+d) as bf16
    gemm(probs, vT, nullptr, attnOut, nullptr, nullptr,
         Nt, HDc, NP, NP, NP, Dm,
         (long long)Hc * SCs, SCs,
         (long long)Hc * HDc * NP, (long long)HDc * NP,
         (long long)Nt * Dm, HDc,
         Zbh, Hc, 0, 1.f, 1);
    // proj
    gemm(attnOut, wbProj + (size_t)i * Dm * Dm, gtmp, nullptr,
         proj_b + i * Dm, nullptr,
         Mtok, Dm, Dm, Dm, Dm, Dm, 0, 0, 0, 0, 0, 0, 1, 1, 0, 1.f, 0);
    residual_kernel<<<ewB, 256, 0, stream>>>(gtmp, cont, hbuf, ybuf);
    layernorm_bf16<<<Mtok, 256, 0, stream>>>(ybuf, Dm, ln2_s + i * Dm, ln2_b + i * Dm,
                                             cont, lnbuf);
    // MLP1 + GELU -> bf16
    gemm(lnbuf, wbM1 + (size_t)i * MLPc * Dm, nullptr, mlpBuf,
         mlp_b1 + i * MLPc, nullptr,
         Mtok, MLPc, Dm, Dm, Dm, MLPc, 0, 0, 0, 0, 0, 0, 1, 1, 0, 1.f, 2);
    // MLP2 -> fp32
    gemm(mlpBuf, wbM2 + (size_t)i * Dm * MLPc, gtmp, nullptr,
         mlp_b2 + i * Dm, nullptr,
         Mtok, Dm, MLPc, MLPc, MLPc, Dm, 0, 0, 0, 0, 0, 0, 1, 1, 0, 1.f, 0);
    act_update<<<Mtok, 256, 0, stream>>>(gtmp, ybuf, hbuf, outAcc,
                                         cumul, rem, cont, (i == Lc - 1) ? 1 : 0);
  }

  // ---- final LN (cls rows only) + head ----
  layernorm_bf16<<<Bc, 256, 0, stream>>>(outAcc, (long long)Nt * Dm,
                                         lnf_s, lnf_b, nullptr, clsBuf);
  gemm(clsBuf, wbHead, logits, nullptr, head_b, nullptr,
       Bc, NCc, Dm, Dm, Dm, NCc, 0, 0, 0, 0, 0, 0, 1, 1, 0, 1.f, 0);
}
